// GATLayer_11811160064008
// MI455X (gfx1250) — compile-verified, matrix operand order checked by reading, not run on hardware
//
#include <hip/hip_runtime.h>
#include <hip/hip_bf16.h>

typedef float    v2f  __attribute__((ext_vector_type(2)));
typedef float    v8f  __attribute__((ext_vector_type(8)));
typedef _Float16 v8h  __attribute__((ext_vector_type(8)));
typedef _Float16 v16h __attribute__((ext_vector_type(16)));
typedef unsigned int v4u __attribute__((ext_vector_type(4)));
typedef int      v4i  __attribute__((ext_vector_type(4)));
typedef int      v8i  __attribute__((ext_vector_type(8)));

#define B_   4
#define N_   2048
#define FIN  128
#define H_   4
#define D_   64
#define HD   256
#define NEG_BIG (-1.0e30f)

// workspace layout (bytes)
//   Vt  : [B*H][D][N] f16  = 16*64*2048*2 = 4 MiB
//   hlT : [B*H][N] f32     = 128 KiB
//   hrT : [B*H][N] f32     = 128 KiB
#define VT_BYTES (16u*64u*2048u*2u)
#define HL_OFF   (VT_BYTES)
#define HR_OFF   (HL_OFF + 16u*2048u*4u)

__global__ void gat_zero_kernel(float* p, int n) {
  int i = blockIdx.x * blockDim.x + threadIdx.x;
  if (i < n) p[i] = 0.0f;
}

// h_prime = h @ W^T via v_wmma_f32_16x16x4_f32; emit Vt (f16, [bh][d][n]) and hl/hr.
__global__ void gat_gemm_kernel(const float* __restrict__ h,
                                const float* __restrict__ W,
                                const float* __restrict__ a,
                                _Float16* __restrict__ Vt,
                                float* __restrict__ hlT,
                                float* __restrict__ hrT) {
  const int lane = threadIdx.x & 31;
  const int wave = threadIdx.x >> 5;
  const int tile = blockIdx.x * 8 + wave;   // 8192 tiles = 512 row-tiles x 16 col-tiles
  const int mt = tile >> 4;
  const int ot = tile & 15;
  const int g0 = mt * 16;                   // global row base (over B*N)
  const int o0 = ot * 16;                   // output-column base (over H*D)
  const int hi = lane >> 4;
  const int l15 = lane & 15;
  const int g = g0 + l15;                   // A row held by this lane
  const int o = o0 + l15;                   // B column held by this lane

  const float* hrow = h + (size_t)g * FIN;
  const float* wrow = W + (size_t)o * FIN;

  // 32-bit A 16x4 layout: lanes 0-15 hold K={0,1}, lanes 16-31 K={2,3} (contiguous pair)
  v8f c = {};
#pragma unroll
  for (int t = 0; t < 32; ++t) {
    v2f av = *(const v2f*)(hrow + 4 * t + 2 * hi);
    v2f bv = *(const v2f*)(wrow + 4 * t + 2 * hi);
    c = __builtin_amdgcn_wmma_f32_16x16x4_f32(false, av, false, bv, (short)0, c,
                                              false, false);
  }

  const int hh = o0 >> 6;                   // head, uniform in wave
  const int d  = o & 63;                    // per-lane dim within head
  const float al = a[hh * 2 * D_ + d];
  const float ar = a[hh * 2 * D_ + D_ + d];
  const int b  = g0 >> 11;                  // batch, uniform in wave (16 | 2048)
  const int bh = b * H_ + hh;

  _Float16* vtb = Vt + (size_t)bh * (D_ * N_);
#pragma unroll
  for (int r = 0; r < 8; ++r) {
    const int m = r + 8 * hi;               // local row in 16x16 C tile
    const int nrow = (g0 + m) & (N_ - 1);   // batch-local row
    float cv = c[r];
    vtb[(size_t)d * N_ + nrow] = (_Float16)cv;   // transposed f16 V
    float vl = cv * al;
    float vr = cv * ar;
#pragma unroll
    for (int off = 1; off < 16; off <<= 1) {     // reduce within each 16-lane half
      vl += __shfl_xor(vl, off, 32);
      vr += __shfl_xor(vr, off, 32);
    }
    if (l15 == 0) {
      atomicAdd(hlT + bh * N_ + nrow, vl);
      atomicAdd(hrT + bh * N_ + nrow, vr);
    }
  }
}

#define LD4I(dst, ptr)                                                         \
  { int4 _t = *(const int4*)(ptr);                                             \
    (dst)[0] = _t.x; (dst)[1] = _t.y; (dst)[2] = _t.z; (dst)[3] = _t.w; }
#define LD4F(dst, ptr)                                                         \
  { float4 _t = *(const float4*)(ptr);                                         \
    (dst)[0] = _t.x; (dst)[1] = _t.y; (dst)[2] = _t.z; (dst)[3] = _t.w; }

// TDM: load one 64(d) x 32(j) f16 tile of Vt (row stride N_ halves) into LDS.
// D# packing per CDNA5 ISA 8.3/8.4: group0 = {count/type, lds_addr, global_addr},
// group1 = {data_size, tensor dims, tile dims, dim0 stride}; groups 2/3 unused (2D).
// amdgpu-toolchain (clang-23) 6-arg builtin form: (v4u, v8i, v4i, v4i, v8i, i32 cpol).
__device__ __forceinline__ void tdm_load_vtile(const _Float16* gsrc, unsigned lds_off) {
  unsigned long long ga = (unsigned long long)(uintptr_t)gsrc;
  v4u g0;
  g0.x = 1u;                                          // count=1, user descriptor
  g0.y = lds_off;                                     // LDS byte address
  g0.z = (unsigned)(ga & 0xffffffffull);              // global_addr[31:0]
  g0.w = (unsigned)((ga >> 32) & 0x01ffffffull)       // global_addr[56:32]
         | 0x80000000u;                               // type=2 ("image")
  v8i g1;
  g1[0] = (int)(1u << 16);       // workgroup_mask=0, data_size=1 (2 bytes)
  g1[1] = (int)((unsigned)N_ << 16);  // tensor_dim0[15:0] = 2048
  g1[2] = (int)(64u << 16);      // tensor_dim0 hi=0, tensor_dim1[15:0] = 64
  g1[3] = (int)(32u << 16);      // tensor_dim1 hi=0, tile_dim0 = 32
  g1[4] = (int)64;               // tile_dim1 = 64, tile_dim2 = 0
  g1[5] = (int)N_;               // tensor_dim0_stride[31:0] = 2048 elements
  g1[6] = 0;                     // stride hi, tensor_dim1_stride lo
  g1[7] = 0;
  v4i gz4 = {0, 0, 0, 0};
  v8i gz8 = {0, 0, 0, 0, 0, 0, 0, 0};
  __builtin_amdgcn_tensor_load_to_lds(g0, g1, gz4, gz4, gz8, 0);
}

// Fused masked-softmax + P@V (flash style). One wave = 16 rows of one (b,h).
// V tiles are staged block-wide in LDS by the Tensor Data Mover, double-buffered.
__global__ void gat_attn_kernel(const int* __restrict__ adj,
                                const _Float16* __restrict__ Vt,
                                const float* __restrict__ hlT,
                                const float* __restrict__ hrT,
                                float* __restrict__ out) {
  const int lane = threadIdx.x & 31;
  const int wave = threadIdx.x >> 5;
  const int bh = blockIdx.x;                // 16 = B*H
  const int b  = bh >> 2;
  const int hh = bh & 3;
  const int i0 = (blockIdx.y * 8 + wave) * 16;
  const int hi = lane >> 4;
  const int l15 = lane & 15;
  const int ig = i0 + l15;                  // score row owned by this lane (both halves)

  __shared__ _Float16 vbuf[2][D_ * 32];     // double-buffered 64x32 V tile

  const float hl = hlT[bh * N_ + ig];
  const float* hrrow = hrT + bh * N_;
  const int* adjrow = adj + ((size_t)b * N_ + ig) * N_;
  const _Float16* vslice = Vt + (size_t)bh * (D_ * N_);

  if (wave == 0) {
    tdm_load_vtile(vslice, (unsigned)(uintptr_t)&vbuf[0][0]);
  }

  v8f acc0 = {}, acc1 = {}, acc2 = {}, acc3 = {};
  float mrun = -3.0e38f;
  float lrun = 0.0f;

  const int vofs = l15 * 32 + 8 * hi;       // per-lane base into a V tile row

  for (int jt = 0; jt < N_ / 32; ++jt) {
    const int cur = jt & 1;
    const int j0 = jt * 32;
    const int kb0 = j0 + 8 * hi;            // first 8-wide K run for this lane
    const int kb1 = j0 + 16 + 8 * hi;       // second 8-wide K run

    __syncthreads();                        // prev reads of buf[cur^1] are done
    if (wave == 0) {
      if (jt + 1 < N_ / 32) {
        tdm_load_vtile(vslice + (jt + 1) * 32, (unsigned)(uintptr_t)&vbuf[cur ^ 1][0]);
        __builtin_amdgcn_s_wait_tensorcnt(1);   // in-order: tile jt has landed
      } else {
        __builtin_amdgcn_s_wait_tensorcnt(0);
      }
    }
    __syncthreads();                        // tile jt visible to whole block

    __builtin_prefetch(adjrow + j0 + 32, 0, 1);   // global_prefetch_b8 next tile

    int adjv[16];
    float hrv[16];
    LD4I(adjv + 0,  adjrow + kb0);
    LD4I(adjv + 4,  adjrow + kb0 + 4);
    LD4I(adjv + 8,  adjrow + kb1);
    LD4I(adjv + 12, adjrow + kb1 + 4);
    LD4F(hrv + 0,  hrrow + kb0);
    LD4F(hrv + 4,  hrrow + kb0 + 4);
    LD4F(hrv + 8,  hrrow + kb1);
    LD4F(hrv + 12, hrrow + kb1 + 4);

    float s[16];
    bool ok[16];
    float tmax = NEG_BIG;
#pragma unroll
    for (int e = 0; e < 16; ++e) {
      const int j = (e < 8) ? (kb0 + e) : (kb1 + (e - 8));
      float x = hl + hrv[e];
      x = (x > 0.0f) ? x : 0.2f * x;        // leaky_relu(0.2)
      ok[e] = (adjv[e] > 0) || (j == ig);   // adj mask OR diagonal
      s[e] = x;
      tmax = fmaxf(tmax, ok[e] ? x : NEG_BIG);
    }
    tmax = fmaxf(tmax, __shfl_xor(tmax, 16, 32));   // combine K-halves of the row
    const float mnew = fmaxf(mrun, tmax);
    const float alpha = __expf(mrun - mnew);
    mrun = mnew;

    float rs = 0.0f;
    v16h A;                                  // P tile, already in WMMA A layout
#pragma unroll
    for (int e = 0; e < 16; ++e) {
      float p = ok[e] ? __expf(s[e] - mnew) : 0.0f;
      rs += p;
      A[e] = (_Float16)p;
    }
    rs += __shfl_xor(rs, 16, 32);
    lrun = lrun * alpha + rs;

#pragma unroll
    for (int r = 0; r < 8; ++r) {            // per-row rescale: row of C-VGPR r is r+8*hi
      const float arr = __shfl(alpha, r + 8 * hi, 32);
      acc0[r] *= arr; acc1[r] *= arr; acc2[r] *= arr; acc3[r] *= arr;
    }

    const _Float16* vb = &vbuf[cur][0];
    v8h blo, bhi2;
    v16h Bv;
    blo  = *(const v8h*)(vb + 0 * 512 + vofs);
    bhi2 = *(const v8h*)(vb + 0 * 512 + vofs + 16);
    Bv = __builtin_shufflevector(blo, bhi2, 0,1,2,3,4,5,6,7,8,9,10,11,12,13,14,15);
    acc0 = __builtin_amdgcn_wmma_f32_16x16x32_f16(false, A, false, Bv, (short)0, acc0, false, false);
    blo  = *(const v8h*)(vb + 1 * 512 + vofs);
    bhi2 = *(const v8h*)(vb + 1 * 512 + vofs + 16);
    Bv = __builtin_shufflevector(blo, bhi2, 0,1,2,3,4,5,6,7,8,9,10,11,12,13,14,15);
    acc1 = __builtin_amdgcn_wmma_f32_16x16x32_f16(false, A, false, Bv, (short)0, acc1, false, false);
    blo  = *(const v8h*)(vb + 2 * 512 + vofs);
    bhi2 = *(const v8h*)(vb + 2 * 512 + vofs + 16);
    Bv = __builtin_shufflevector(blo, bhi2, 0,1,2,3,4,5,6,7,8,9,10,11,12,13,14,15);
    acc2 = __builtin_amdgcn_wmma_f32_16x16x32_f16(false, A, false, Bv, (short)0, acc2, false, false);
    blo  = *(const v8h*)(vb + 3 * 512 + vofs);
    bhi2 = *(const v8h*)(vb + 3 * 512 + vofs + 16);
    Bv = __builtin_shufflevector(blo, bhi2, 0,1,2,3,4,5,6,7,8,9,10,11,12,13,14,15);
    acc3 = __builtin_amdgcn_wmma_f32_16x16x32_f16(false, A, false, Bv, (short)0, acc3, false, false);
  }

  const float inv = 1.0f / lrun;
#pragma unroll
  for (int r = 0; r < 8; ++r) {
    const float ir = __shfl(inv, r + 8 * hi, 32);
    const int irow = i0 + r + 8 * hi;
    float* po = out + ((size_t)b * N_ + irow) * HD + hh * D_;
    po[0 * 16 + l15] = acc0[r] * ir;
    po[1 * 16 + l15] = acc1[r] * ir;
    po[2 * 16 + l15] = acc2[r] * ir;
    po[3 * 16 + l15] = acc3[r] * ir;
  }
}

extern "C" void kernel_launch(void* const* d_in, const int* in_sizes, int n_in,
                              void* d_out, int out_size, void* d_ws, size_t ws_size,
                              hipStream_t stream) {
  (void)in_sizes; (void)n_in; (void)out_size; (void)ws_size;
  const float* h   = (const float*)d_in[0];
  const int*   adj = (const int*)d_in[1];
  const float* W   = (const float*)d_in[2];
  const float* a   = (const float*)d_in[3];
  float* out = (float*)d_out;

  _Float16* Vt = (_Float16*)d_ws;
  float* hlT = (float*)((char*)d_ws + HL_OFF);
  float* hrT = (float*)((char*)d_ws + HR_OFF);

  // hlT and hrT are contiguous: zero both with one launch (2 * 16 * 2048 floats)
  gat_zero_kernel<<<dim3((2 * 16 * N_ + 255) / 256), dim3(256), 0, stream>>>(hlT, 2 * 16 * N_);
  gat_gemm_kernel<<<dim3(1024), dim3(256), 0, stream>>>(h, W, a, Vt, hlT, hrT);
  gat_attn_kernel<<<dim3(16, 16), dim3(256), 0, stream>>>(adj, Vt, hlT, hrT, out);
}
